// MultiModalAttention_50912542327029
// MI455X (gfx1250) — matrix operand order repaired and checked by reading, not compiled, not run on hardware
//
#include <hip/hip_runtime.h>

#define DEV __device__ __forceinline__

typedef __attribute__((ext_vector_type(16))) __bf16   v16bf;
typedef __attribute__((ext_vector_type(8)))  float    v8f;
typedef __attribute__((ext_vector_type(4)))  unsigned u32x4;
typedef __attribute__((ext_vector_type(4)))  float    f32x4;
typedef __attribute__((ext_vector_type(2)))  float    f32x2;

union Frag { u32x4 q[2]; v16bf v; };

static constexpr int SQ = 512, Hh = 16, Dm = 1024;

// fp32 -> bf16 via native conversion (RNE); pairs pack into one dword so the
// compiler can use v_cvt_pk_bf16_f32.
DEV unsigned short f2bf(float f) {
    return __builtin_bit_cast(unsigned short, (__bf16)f);
}
DEV unsigned pk2bf(float a, float b) {
    unsigned short ua = __builtin_bit_cast(unsigned short, (__bf16)a);
    unsigned short ub = __builtin_bit_cast(unsigned short, (__bf16)b);
    return (unsigned)ua | ((unsigned)ub << 16);
}

// A-matrix 16x32 bf16 fragment: lane 0-15 row M=lane, K=0..15;
// lanes 16-31 same rows, K=16..31.
DEV v16bf load_a_frag(const unsigned short* base, int stride, int lane) {
    int row = lane & 15;
    int kb  = (lane >> 4) << 4;
    const unsigned short* p = base + (size_t)row * stride;
    Frag f;
    f.q[0] = *(const u32x4*)(p + kb);
    f.q[1] = *(const u32x4*)(p + kb + 8);
    return f.v;
}

// B-matrix 32x16 bf16 fragment: lane 0-15 col N=lane, K=0..15; lanes 16-31 K=16..31.
DEV v16bf load_b_frag(const unsigned short* base, int stride, int lane) {
    int n  = lane & 15;
    int kb = (lane >> 4) << 4;
    const unsigned short* p = base + (size_t)n * stride + kb;
    Frag f;
    f.q[0] = *(const u32x4*)(p);
    f.q[1] = *(const u32x4*)(p + 8);
    return f.v;
}

DEV v8f wmma_bf16(v16bf a, v16bf b, v8f c) {
    return __builtin_amdgcn_wmma_f32_16x16x32_bf16(false, a, false, b, (short)0, c,
                                                   false, false);
}

// ---------------------------------------------------------------------------
// Generic projection GEMM: C(M x 1024) = A(f32) * W(f32, 1024x1024) + bias.
// mode 0: bf16 out, (B,H,S,DK) layout     (Q, K tensors)
// mode 1: bf16 out, (B,H,DK,S) layout     (V tensors, pre-transposed for PV)
// mode 2: f32 out, row-major M x 1024     (final output projection)
// ---------------------------------------------------------------------------
__global__ __launch_bounds__(128)
void gemm_proj_kernel(const float* __restrict__ A, const float* __restrict__ W,
                      const float* __restrict__ bias, void* __restrict__ out,
                      int M, int S, int mode) {
    __shared__ alignas(16) unsigned short As[128][48];
    __shared__ alignas(16) unsigned short Bs[128][48];
    __shared__ alignas(16) unsigned short Tt[4][64][72];  // mode-1 transpose staging

    const int t    = threadIdx.x;
    const int lane = t & 31;
    const int wid  = t >> 5;
    const int wm   = wid >> 1;     // 0..1
    const int wn   = wid & 1;      // 0..1
    const int m0   = blockIdx.x * 128;
    const int n0   = blockIdx.y * 128;

    v8f acc[4][4];
    v8f zero = {};
    #pragma unroll
    for (int i = 0; i < 4; ++i)
        #pragma unroll
        for (int j = 0; j < 4; ++j) acc[i][j] = zero;

    const float* ap = A + (size_t)(m0 + t) * Dm;          // A row t of tile
    const float* wp = W + n0 + t;                          // W column t of tile

    for (int k0 = 0; k0 < Dm; k0 += 32) {
        // Stage A tile 128x32: pack 8 f32 -> 4 dwords -> one b128 store.
        #pragma unroll
        for (int i = 0; i < 4; ++i) {
            f32x4 v0 = *(const f32x4*)(ap + k0 + i * 8);
            f32x4 v1 = *(const f32x4*)(ap + k0 + i * 8 + 4);
            u32x4 p;
            p[0] = pk2bf(v0[0], v0[1]);
            p[1] = pk2bf(v0[2], v0[3]);
            p[2] = pk2bf(v1[0], v1[1]);
            p[3] = pk2bf(v1[2], v1[3]);
            *(u32x4*)(&As[t][i * 8]) = p;
        }
        // Stage W tile 32x128 transposed: thread t owns column n = t; reads are
        // coalesced across lanes (consecutive lanes -> consecutive addresses).
        #pragma unroll
        for (int i = 0; i < 4; ++i) {
            float x[8];
            #pragma unroll
            for (int e = 0; e < 8; ++e)
                x[e] = wp[(size_t)(k0 + i * 8 + e) * Dm];
            u32x4 p;
            p[0] = pk2bf(x[0], x[1]);
            p[1] = pk2bf(x[2], x[3]);
            p[2] = pk2bf(x[4], x[5]);
            p[3] = pk2bf(x[6], x[7]);
            *(u32x4*)(&Bs[t][i * 8]) = p;
        }
        if (k0 + 32 < Dm) {                        // prefetch next K-step
            __builtin_prefetch(ap + k0 + 32, 0, 1);
            __builtin_prefetch(wp + (size_t)(k0 + 32) * Dm, 0, 1);
        }
        __syncthreads();

        v16bf af[4], bfr[4];
        #pragma unroll
        for (int mi = 0; mi < 4; ++mi)
            af[mi] = load_a_frag(&As[wm * 64 + mi * 16][0], 48, lane);
        #pragma unroll
        for (int ni = 0; ni < 4; ++ni)
            bfr[ni] = load_b_frag(&Bs[wn * 64 + ni * 16][0], 48, lane);
        #pragma unroll
        for (int mi = 0; mi < 4; ++mi)
            #pragma unroll
            for (int ni = 0; ni < 4; ++ni)
                acc[mi][ni] = wmma_bf16(af[mi], bfr[ni], acc[mi][ni]);
        __syncthreads();
    }

    // Epilogue. C layout (ISA 7.12.2): VGPR j, lane -> M = j + 8*(lane>>4), N = lane&15.
    const int half = lane >> 4;
    const int bidx = m0 / S;              // tile never crosses a batch (S % 128 == 0)
    const int s0   = m0 - bidx * S;

    if (mode == 1) {
        // Transpose each wave's 64x64 tile through LDS, then 16B-contiguous stores.
        #pragma unroll
        for (int mi = 0; mi < 4; ++mi)
            #pragma unroll
            for (int ni = 0; ni < 4; ++ni)
                #pragma unroll
                for (int j = 0; j < 8; ++j) {
                    int mwl = mi * 16 + j + half * 8;
                    int nwl = ni * 16 + (lane & 15);
                    int gn  = n0 + wn * 64 + nwl;
                    Tt[wid][nwl][mwl] = f2bf(acc[mi][ni][j] + bias[gn]);
                }
        __syncthreads();
        unsigned short* o16 = (unsigned short*)out;
        for (int n = lane; n < 64; n += 32) {
            int gn = n0 + wn * 64 + n;
            int h = gn >> 6, dk = gn & 63;
            unsigned short* op =
                o16 + ((size_t)(bidx * Hh + h) * 64 + dk) * S + s0 + wm * 64;
            #pragma unroll
            for (int c = 0; c < 8; ++c)
                *(u32x4*)(op + c * 8) = *(u32x4*)(&Tt[wid][n][c * 8]);
        }
        return;
    }

    #pragma unroll
    for (int mi = 0; mi < 4; ++mi) {
        #pragma unroll
        for (int ni = 0; ni < 4; ++ni) {
            #pragma unroll
            for (int j = 0; j < 8; ++j) {
                int ml = wm * 64 + mi * 16 + j + half * 8;
                int nl = wn * 64 + ni * 16 + (lane & 15);
                int gm = m0 + ml, gn = n0 + nl;
                float val = acc[mi][ni][j] + bias[gn];
                if (mode == 2) {
                    ((float*)out)[(size_t)gm * Dm + gn] = val;
                } else {
                    int s = ml + s0 + (wm ? 0 : 0);  // s = gm - bidx*S
                    s = gm - bidx * S;
                    int h = gn >> 6, dk = gn & 63;
                    ((unsigned short*)out)[((size_t)(bidx * Hh + h) * S + s) * 64 + dk] =
                        f2bf(val);
                }
            }
        }
    }
}

// ---------------------------------------------------------------------------
// One attention branch. Workgroup = 128 threads (4 waves) = 64 query rows of
// one (b,h). Full score rows live in LDS (<=306KB, fits the 320KB WGP LDS).
//   qh : (B,H,SQ,64) bf16, kh : (B,H,Sk,64) bf16, vtr : (B,H,64,Sk) bf16
//   comb accumulates (B,SQ,1024) f32 "concat" layout.
// ---------------------------------------------------------------------------
__global__ __launch_bounds__(128)
void attn_kernel(const unsigned short* __restrict__ qh,
                 const unsigned short* __restrict__ kh,
                 const unsigned short* __restrict__ vtr,
                 const int* __restrict__ mask, const int* __restrict__ pad,
                 float* __restrict__ aw, float* __restrict__ comb,
                 float wgt, int Sk, int first) {
    extern __shared__ char smem[];
    unsigned short* Qs  = (unsigned short*)smem;                 // 64 x 72 bf16
    float*          inv = (float*)(smem + 64 * 72 * 2);          // 64 f32
    float*          Sc  = (float*)(smem + 9472);                 // 64 x (Sk+1) f32
    unsigned short* Pb  =
        (unsigned short*)(smem + 9472 + (size_t)64 * (Sk + 1) * 4); // 64 x (Sk+8)

    const int t    = threadIdx.x;
    const int lane = t & 31;
    const int wid  = t >> 5;
    const int q0   = blockIdx.x * 64;
    const int bh   = blockIdx.y;
    const int b    = bh >> 4;
    const int h    = bh & 15;
    const int sst  = Sk + 1;
    const int pst  = Sk + 8;
    const int half = lane >> 4;

    // Load 64x64 Q tile.
    {
        int r = t >> 1, c = (t & 1) * 32;
        const unsigned short* src = qh + ((size_t)bh * SQ + q0 + r) * 64 + c;
        u32x4* d = (u32x4*)&Qs[r * 72 + c];
        d[0] = ((const u32x4*)src)[0];
        d[1] = ((const u32x4*)src)[1];
    }
    __syncthreads();

    // Phase 1: S = Q K^T / 8, masked, into LDS. Wave w owns rows 16w..16w+15.
    const unsigned short* kbase = kh + (size_t)bh * Sk * 64;
    for (int n0 = 0; n0 < Sk; n0 += 16) {
        v8f acc = {};
        #pragma unroll
        for (int ks = 0; ks < 64; ks += 32) {
            v16bf a   = load_a_frag(&Qs[(wid * 16) * 72 + ks], 72, lane);
            v16bf bfr = load_b_frag(kbase + (size_t)n0 * 64 + ks, 64, lane);
            acc = wmma_bf16(a, bfr, acc);
        }
        int col = n0 + (lane & 15);
        #pragma unroll
        for (int j = 0; j < 8; ++j) {
            int ml = wid * 16 + j + half * 8;
            float s = acc[j] * 0.125f;   // 1/sqrt(64)
            int gq = q0 + ml;
            if (mask && mask[((size_t)b * SQ + gq) * Sk + col]) s = -1e30f;
            if (pad && pad[(size_t)b * Sk + col]) s = -1e30f;
            Sc[ml * sst + col] = s;
        }
    }
    __syncthreads();

    // Phase 2a: per-row max / exp / sum.
    if (t < 64) {
        float* row = &Sc[t * sst];
        float mx = -3.4e38f;
        for (int c = 0; c < Sk; ++c) mx = fmaxf(mx, row[c]);
        float sum = 0.f;
        for (int c = 0; c < Sk; ++c) {
            float e = __expf(row[c] - mx);
            row[c] = e;
            sum += e;
        }
        inv[t] = 1.0f / sum;
    }
    __syncthreads();

    // Phase 2b: normalize pairs; one b32 prob store + one 8B aw store per pair.
    const int hSk = Sk >> 1;
    for (int idx = t; idx < 64 * hSk; idx += 128) {
        int m = idx / hSk, c2 = (idx - m * hSk) * 2;
        float w0 = Sc[m * sst + c2] * inv[m];
        float w1 = Sc[m * sst + c2 + 1] * inv[m];
        *(unsigned*)&Pb[m * pst + c2] = pk2bf(w0, w1);
        if (aw) {
            f32x2 wv; wv[0] = w0; wv[1] = w1;
            *(f32x2*)(aw + ((size_t)bh * SQ + q0 + m) * Sk + c2) = wv;
        }
    }
    __syncthreads();

    // Phase 3: O = P * V (V pre-transposed -> contiguous B-fragment loads).
    const unsigned short* vbase = vtr + (size_t)bh * 64 * Sk;
    v8f oacc[4];
    v8f zero = {};
    #pragma unroll
    for (int ni = 0; ni < 4; ++ni) oacc[ni] = zero;
    for (int k0 = 0; k0 < Sk; k0 += 32) {
        v16bf a = load_a_frag(&Pb[(wid * 16) * pst + k0], pst, lane);
        #pragma unroll
        for (int ni = 0; ni < 4; ++ni) {
            v16bf bfr = load_b_frag(vbase + (size_t)(ni * 16) * Sk + k0, Sk, lane);
            oacc[ni] = wmma_bf16(a, bfr, oacc[ni]);
        }
    }

    // Accumulate weighted branch output into concat (B,SQ,1024) f32 buffer.
    #pragma unroll
    for (int ni = 0; ni < 4; ++ni) {
        #pragma unroll
        for (int j = 0; j < 8; ++j) {
            int ml = wid * 16 + j + half * 8;
            int dk = ni * 16 + (lane & 15);
            size_t o = ((size_t)b * SQ + q0 + ml) * Dm + h * 64 + dk;
            float v = oacc[ni][j] * wgt;
            comb[o] = first ? v : (comb[o] + v);
        }
    }
}

// ---------------------------------------------------------------------------
extern "C" void kernel_launch(void* const* d_in, const int* in_sizes, int n_in,
                              void* d_out, int out_size, void* d_ws, size_t ws_size,
                              hipStream_t stream) {
    const float* q    = (const float*)d_in[0];
    const float* k_e  = (const float*)d_in[1];
    const float* v_e  = (const float*)d_in[2];
    const float* k_i  = (const float*)d_in[3];
    const float* v_i  = (const float*)d_in[4];
    const float* k_ei = (const float*)d_in[5];
    const float* v_ei = (const float*)d_in[6];
    const int* mask_e  = (const int*)d_in[7];
    const int* mask_ei = (const int*)d_in[8];
    const int* pad_e   = (const int*)d_in[9];
    const int* pad_ei  = (const int*)d_in[10];
    const float* Wq   = (const float*)d_in[11], *bq   = (const float*)d_in[12];
    const float* Wke  = (const float*)d_in[13], *bke  = (const float*)d_in[14];
    const float* Wve  = (const float*)d_in[15], *bve  = (const float*)d_in[16];
    const float* Wki  = (const float*)d_in[17], *bki  = (const float*)d_in[18];
    const float* Wvi  = (const float*)d_in[19], *bvi  = (const float*)d_in[20];
    const float* Wkei = (const float*)d_in[21], *bkei = (const float*)d_in[22];
    const float* Wvei = (const float*)d_in[23], *bvei = (const float*)d_in[24];
    const float* Wout = (const float*)d_in[25], *bout = (const float*)d_in[26];

    char* ws = (char*)d_ws;
    const size_t szE  = (size_t)16 * 16 * 512 * 64 * 2;  // bf16 bytes, S=512
    const size_t szI  = (size_t)16 * 16 * 256 * 64 * 2;
    const size_t szEI = (size_t)16 * 16 * 768 * 64 * 2;
    unsigned short* qh   = (unsigned short*)(ws);
    unsigned short* keh  = (unsigned short*)(ws + szE);
    unsigned short* veh  = (unsigned short*)(ws + 2 * szE);
    unsigned short* kih  = (unsigned short*)(ws + 3 * szE);
    unsigned short* vih  = (unsigned short*)(ws + 3 * szE + szI);
    unsigned short* keih = (unsigned short*)(ws + 3 * szE + 2 * szI);
    unsigned short* veih = (unsigned short*)(ws + 3 * szE + 2 * szI + szEI);
    float*          comb = (float*)(ws + 3 * szE + 2 * szI + 2 * szEI);

    float* out  = (float*)d_out;                      // (16,512,1024)
    float* aw_e = out + (size_t)16 * 512 * 1024;      // (16,16,512,512)
    float* aw_i = aw_e + (size_t)16 * 16 * 512 * 512; // (16,16,512,256)

    dim3 blk(128);

    // Projections (M = B*S rows, 8 N-tiles of 128).
    gemm_proj_kernel<<<dim3(64, 8), blk, 0, stream>>>(q,    Wq,   bq,   qh,   8192,  512, 0);
    gemm_proj_kernel<<<dim3(64, 8), blk, 0, stream>>>(k_e,  Wke,  bke,  keh,  8192,  512, 0);
    gemm_proj_kernel<<<dim3(64, 8), blk, 0, stream>>>(v_e,  Wve,  bve,  veh,  8192,  512, 1);
    gemm_proj_kernel<<<dim3(32, 8), blk, 0, stream>>>(k_i,  Wki,  bki,  kih,  4096,  256, 0);
    gemm_proj_kernel<<<dim3(32, 8), blk, 0, stream>>>(v_i,  Wvi,  bvi,  vih,  4096,  256, 1);
    gemm_proj_kernel<<<dim3(96, 8), blk, 0, stream>>>(k_ei, Wkei, bkei, keih, 12288, 768, 0);
    gemm_proj_kernel<<<dim3(96, 8), blk, 0, stream>>>(v_ei, Wvei, bvei, veih, 12288, 768, 1);

    auto smem = [](int Sk) {
        return (size_t)9472 + (size_t)64 * (Sk + 1) * 4 + (size_t)64 * (Sk + 8) * 2;
    };
    // Three attention branches, stream-ordered accumulation into comb.
    attn_kernel<<<dim3(8, 256), blk, smem(512), stream>>>(qh, keh, veh, mask_e, pad_e,
                                                          aw_e, comb, 1.0f, 512, 1);
    attn_kernel<<<dim3(8, 256), blk, smem(256), stream>>>(qh, kih, vih, nullptr, nullptr,
                                                          aw_i, comb, 0.5f, 256, 0);
    attn_kernel<<<dim3(8, 256), blk, smem(768), stream>>>(qh, keih, veih, mask_ei, pad_ei,
                                                          nullptr, comb, 0.5f, 768, 0);

    // Output projection: concat (f32) @ Wout + bout -> d_out (f32).
    gemm_proj_kernel<<<dim3(64, 8), blk, 0, stream>>>(comb, Wout, bout, (void*)out,
                                                      8192, 512, 2);
}
